// FlattenedObsEncoder_30090540876219
// MI455X (gfx1250) — matrix operand order repaired and checked by reading, not compiled, not run on hardware
//
#include <hip/hip_runtime.h>

// ---------------------------------------------------------------------------
// FlattenedObsEncoder for MI455X (gfx1250, wave32, WMMA)
//   out[b,n] = sum_k (W_embed[x[b, k/4], k%4] + b_embed[k%4]) * W_dense[k,n] + b_dense[n]
//   B=16384, K=1024 (256 tokens * 4 channels), N=1024, classes=1024
// Strategy: bf16x3 split-precision GEMM on v_wmma_f32_16x16x32_bf16, with
// double-buffered async Global->LDS staging (ASYNCcnt) of the W_dense slices.
// ---------------------------------------------------------------------------

#define B_DIM 16384
#define TOK   256
#define EMBED 4
#define NCLS  1024
#define K_DIM 1024
#define N_DIM 1024

typedef __attribute__((ext_vector_type(16))) __bf16 v16bf;
typedef __attribute__((ext_vector_type(8)))  __bf16 v8bf;
typedef __attribute__((ext_vector_type(4)))  __bf16 v4bf;
typedef __attribute__((ext_vector_type(8)))  float  v8f;

union AFrag { v16bf v; v4bf q[4]; };
union BFrag { v16bf v; v8bf d[2]; };

// ---- CDNA5 async Global->LDS copy (16B per lane), with safe fallback -------
#if defined(__has_builtin)
#if __has_builtin(__builtin_amdgcn_global_load_async_to_lds_b128)
#define USE_ASYNC_STAGE 1
#endif
#endif
#ifndef USE_ASYNC_STAGE
#define USE_ASYNC_STAGE 0
#endif

typedef int v4i_vs __attribute__((vector_size(16)));
typedef __attribute__((address_space(1))) v4i_vs* gptr_v4i;
typedef __attribute__((address_space(3))) v4i_vs* lptr_v4i;

__device__ __forceinline__ void cp16_g2l(void* lds, const void* gmem) {
#if USE_ASYNC_STAGE
    __builtin_amdgcn_global_load_async_to_lds_b128(
        (gptr_v4i)gmem, (lptr_v4i)lds, 0, 0);
#else
    *(uint4*)lds = *(const uint4*)gmem;
#endif
}

template <int N>
__device__ __forceinline__ void wait_async() {
#if USE_ASYNC_STAGE
#if __has_builtin(__builtin_amdgcn_s_wait_asynccnt)
    __builtin_amdgcn_s_wait_asynccnt(N);
#else
    asm volatile("s_wait_asynccnt %0" :: "i"(N) : "memory");
#endif
#endif
}

// round-to-nearest-even f32 -> bf16 bit pattern (pure integer, no __bf16 arith)
__device__ __forceinline__ unsigned short f32_to_bf16_rne(float f) {
    unsigned u = __float_as_uint(f);
    unsigned rounding = 0x7FFFu + ((u >> 16) & 1u);
    return (unsigned short)((u + rounding) >> 16);
}
__device__ __forceinline__ float bf16_bits_to_f32(unsigned short b) {
    return __uint_as_float(((unsigned)b) << 16);
}
__device__ __forceinline__ void split_hi_lo(float w, unsigned short& hi, unsigned short& lo) {
    hi = f32_to_bf16_rne(w);
    lo = f32_to_bf16_rne(w - bf16_bits_to_f32(hi));
}

// --- pre-kernel 1: split W_embed (1024x4) into bf16 hi/lo tables ------------
__global__ void split_embed_kernel(const float* __restrict__ W,
                                   unsigned short* __restrict__ hi,
                                   unsigned short* __restrict__ lo) {
    int i = blockIdx.x * 256 + threadIdx.x;   // 4096 elements
    unsigned short h, l;
    split_hi_lo(W[i], h, l);
    hi[i] = h;
    lo[i] = l;
}

// --- pre-kernel 2: split + transpose W_dense (K x N) -> hi/lo (N x K) -------
__global__ void split_transpose_dense_kernel(const float* __restrict__ W,
                                             unsigned short* __restrict__ hiT,
                                             unsigned short* __restrict__ loT) {
    __shared__ unsigned short th[32][33];
    __shared__ unsigned short tl[32][33];
    const int k0 = blockIdx.y * 32, n0 = blockIdx.x * 32;
    const int tx = threadIdx.x, ty = threadIdx.y;   // (32, 8)
#pragma unroll
    for (int r = 0; r < 4; ++r) {
        int k = ty + r * 8;
        float w = W[(size_t)(k0 + k) * N_DIM + n0 + tx];   // coalesced over n
        unsigned short h, l;
        split_hi_lo(w, h, l);
        th[k][tx] = h;
        tl[k][tx] = l;
    }
    __syncthreads();
#pragma unroll
    for (int r = 0; r < 4; ++r) {
        int n = ty + r * 8;
        hiT[(size_t)(n0 + n) * K_DIM + k0 + tx] = th[tx][n];  // coalesced over k
        loT[(size_t)(n0 + n) * K_DIM + k0 + tx] = tl[tx][n];
    }
}

// --- pre-kernel 3: bias_total[n] = b_dense[n] + sum_k b_embed[k&3]*W_dense[k,n]
__global__ void bias_fold_kernel(const float* __restrict__ Wd,
                                 const float* __restrict__ be,
                                 const float* __restrict__ bd,
                                 float* __restrict__ bias) {
    int n = blockIdx.x * 256 + threadIdx.x;
    float e0 = be[0], e1 = be[1], e2 = be[2], e3 = be[3];
    float acc = bd[n];
    for (int k = 0; k < K_DIM; k += 4) {
        acc += e0 * Wd[(size_t)(k + 0) * N_DIM + n];
        acc += e1 * Wd[(size_t)(k + 1) * N_DIM + n];
        acc += e2 * Wd[(size_t)(k + 2) * N_DIM + n];
        acc += e3 * Wd[(size_t)(k + 3) * N_DIM + n];
    }
    bias[n] = acc;
}

// --- main kernel: gather-A bf16x3 GEMM with async double-buffered staging ---
// Block: 256 threads (8 waves). Block tile: 128 (M) x 128 (N). K step = 32.
// Wave (wm 0..3, wn 0..1) owns 32 (M) x 64 (N): 2x4 accumulator tiles.
#define BPITCH 40   // ushorts per B-tile row in LDS (32 + pad, keeps 16B align)
#define KSTEPS (K_DIM / 32)

__global__ __launch_bounds__(256) void encoder_gemm_kernel(
    const int* __restrict__ x,
    const unsigned short* __restrict__ wd_hi,   // [N][K] bf16 bits
    const unsigned short* __restrict__ wd_lo,
    const unsigned short* __restrict__ emb_hi,  // [NCLS][4] bf16 bits
    const unsigned short* __restrict__ emb_lo,
    const float* __restrict__ bias,             // [N]
    float* __restrict__ out)                    // [B][N]
{
    __shared__ __align__(16) unsigned short s_emb_hi[NCLS * EMBED];     // 8KB
    __shared__ __align__(16) unsigned short s_emb_lo[NCLS * EMBED];     // 8KB
    __shared__ __align__(16) unsigned short s_b_hi[2][128 * BPITCH];    // 20KB
    __shared__ __align__(16) unsigned short s_b_lo[2][128 * BPITCH];    // 20KB
    __shared__ __align__(16) int            s_x[2][128 * 8];            // 8KB

    const int tid  = threadIdx.x;
    const int lane = tid & 31;
    const int wid  = tid >> 5;
    const int h    = lane >> 4;      // lane half: selects K sub-range
    const int ln   = lane & 15;      // M (for A) / N (for B,C,D) position
    const int wm   = wid & 3;        // wave M slot (0..3)
    const int wn   = wid >> 2;       // wave N slot (0..1)
    const int m0   = blockIdx.y * 128;
    const int n0   = blockIdx.x * 128;

    // per-thread staging addresses (constant across K loop except the ks term)
    const int    srow    = tid >> 1;        // 0..127
    const int    shalf   = tid & 1;
    const int    koff    = shalf << 4;      // 0 or 16 (bf16 elements)
    const int    toff    = shalf << 2;      // 0 or 4  (int tokens)
    const size_t gb_base = (size_t)(n0 + srow) * K_DIM + koff;
    const size_t gx_base = (size_t)(m0 + srow) * TOK + toff;
    const int    lb      = srow * BPITCH + koff;
    const int    lx      = srow * 8 + toff;

    // 5 x 16B async copies per thread per K-slice: B-hi, B-lo (32B each), x (16B)
    auto issue_stage = [&](int buf, int ks) {
        const size_t gb = gb_base + (size_t)ks * 32;
        cp16_g2l(&s_b_hi[buf][lb],     wd_hi + gb);
        cp16_g2l(&s_b_hi[buf][lb + 8], wd_hi + gb + 8);
        cp16_g2l(&s_b_lo[buf][lb],     wd_lo + gb);
        cp16_g2l(&s_b_lo[buf][lb + 8], wd_lo + gb + 8);
        cp16_g2l(&s_x[buf][lx],        x + gx_base + (size_t)ks * 8);
    };

    // prime buffer 0 while we do the one-time block setup below
    issue_stage(0, 0);

    // preload embedding split tables into LDS (512 uint4 per table)
    {
        const uint4* gh = (const uint4*)emb_hi;
        const uint4* gl = (const uint4*)emb_lo;
        uint4* dh = (uint4*)s_emb_hi;
        uint4* dl = (uint4*)s_emb_lo;
        dh[tid] = gh[tid];  dh[tid + 256] = gh[tid + 256];
        dl[tid] = gl[tid];  dl[tid + 256] = gl[tid + 256];
    }

    // accumulators seeded with folded bias (same n for all 8 rows of a tile)
    v8f acc[2][4];
#pragma unroll
    for (int nt = 0; nt < 4; ++nt) {
        float bv = bias[n0 + wn * 64 + nt * 16 + ln];
        v8f a;
#pragma unroll
        for (int r = 0; r < 8; ++r) a[r] = bv;
        acc[0][nt] = a;
        acc[1][nt] = a;
    }

    for (int ks = 0; ks < KSTEPS; ++ks) {
        const int cur = ks & 1;
        if (ks + 1 < KSTEPS) {
            // next tile's copies overlap this tile's compute; the buffer they
            // write was last read before the previous trailing barrier.
            issue_stage(cur ^ 1, ks + 1);
            wait_async<5>();   // older (current-tile) copies retired in order
        } else {
            wait_async<0>();
        }
        __syncthreads();       // everyone's current-tile copies landed

        // ---- A fragments: 4 token gathers per lane, channels 0..3 contiguous.
        // 16-bit A layout: e0-3 -> token h*4, e4-7 -> h*4+1, e8-11 -> h*4+2, e12-15 -> h*4+3
        AFrag ahi[2], alo[2];
#pragma unroll
        for (int mt = 0; mt < 2; ++mt) {
            const int* xr = s_x[cur] + (wm * 32 + mt * 16 + ln) * 8;
            const int t0 = h * 4;
            int i0 = xr[t0] * 4, i1 = xr[t0 + 1] * 4;
            int i2 = xr[t0 + 2] * 4, i3 = xr[t0 + 3] * 4;
            ahi[mt].q[0] = *(const v4bf*)(s_emb_hi + i0);
            ahi[mt].q[1] = *(const v4bf*)(s_emb_hi + i1);
            ahi[mt].q[2] = *(const v4bf*)(s_emb_hi + i2);
            ahi[mt].q[3] = *(const v4bf*)(s_emb_hi + i3);
            alo[mt].q[0] = *(const v4bf*)(s_emb_lo + i0);
            alo[mt].q[1] = *(const v4bf*)(s_emb_lo + i1);
            alo[mt].q[2] = *(const v4bf*)(s_emb_lo + i2);
            alo[mt].q[3] = *(const v4bf*)(s_emb_lo + i3);
        }

        // ---- B fragments (16 contiguous K per lane) + 3x WMMA per tile
#pragma unroll
        for (int nt = 0; nt < 4; ++nt) {
            BFrag bhi, blo;
            const unsigned short* bp = s_b_hi[cur] + (wn * 64 + nt * 16 + ln) * BPITCH + h * 16;
            const unsigned short* bq = s_b_lo[cur] + (wn * 64 + nt * 16 + ln) * BPITCH + h * 16;
            bhi.d[0] = *(const v8bf*)bp;  bhi.d[1] = *(const v8bf*)(bp + 8);
            blo.d[0] = *(const v8bf*)bq;  blo.d[1] = *(const v8bf*)(bq + 8);
#pragma unroll
            for (int mt = 0; mt < 2; ++mt) {
                acc[mt][nt] = __builtin_amdgcn_wmma_f32_16x16x32_bf16(
                    false, alo[mt].v, false, bhi.v, (short)0, acc[mt][nt], false, false);
                acc[mt][nt] = __builtin_amdgcn_wmma_f32_16x16x32_bf16(
                    false, ahi[mt].v, false, blo.v, (short)0, acc[mt][nt], false, false);
                acc[mt][nt] = __builtin_amdgcn_wmma_f32_16x16x32_bf16(
                    false, ahi[mt].v, false, bhi.v, (short)0, acc[mt][nt], false, false);
            }
        }
        __syncthreads();   // reads of buf[cur] done before it is refilled
    }

    // ---- epilogue: C/D layout is m = h*8 + r, n = ln within each 16x16 tile
#pragma unroll
    for (int mt = 0; mt < 2; ++mt) {
        const int mb = m0 + wm * 32 + mt * 16 + h * 8;
#pragma unroll
        for (int nt = 0; nt < 4; ++nt) {
            const int n = n0 + wn * 64 + nt * 16 + ln;
#pragma unroll
            for (int r = 0; r < 8; ++r)
                out[(size_t)(mb + r) * N_DIM + n] = acc[mt][nt][r];
        }
    }
}

// ---------------------------------------------------------------------------
extern "C" void kernel_launch(void* const* d_in, const int* in_sizes, int n_in,
                              void* d_out, int out_size, void* d_ws, size_t ws_size,
                              hipStream_t stream) {
    (void)in_sizes; (void)n_in; (void)out_size; (void)ws_size;
    const int*   x       = (const int*)d_in[0];
    const float* W_embed = (const float*)d_in[1];
    const float* b_embed = (const float*)d_in[2];
    const float* W_dense = (const float*)d_in[3];
    const float* b_dense = (const float*)d_in[4];
    float* out = (float*)d_out;

    // workspace layout (needs ~4.02 MB)
    char* ws = (char*)d_ws;
    unsigned short* wd_hi  = (unsigned short*)ws;                             // 2 MB
    unsigned short* wd_lo  = (unsigned short*)(ws + (size_t)2 * 1024 * 1024); // 2 MB
    unsigned short* emb_hi = (unsigned short*)(ws + (size_t)4 * 1024 * 1024);        // 8 KB
    unsigned short* emb_lo = (unsigned short*)(ws + (size_t)4 * 1024 * 1024 + 8192); // 8 KB
    float*          bias   = (float*)(ws + (size_t)4 * 1024 * 1024 + 16384);         // 4 KB

    split_embed_kernel<<<NCLS * EMBED / 256, 256, 0, stream>>>(W_embed, emb_hi, emb_lo);
    split_transpose_dense_kernel<<<dim3(N_DIM / 32, K_DIM / 32), dim3(32, 8), 0, stream>>>(
        W_dense, wd_hi, wd_lo);
    bias_fold_kernel<<<N_DIM / 256, 256, 0, stream>>>(W_dense, b_embed, b_dense, bias);

    encoder_gemm_kernel<<<dim3(N_DIM / 128, B_DIM / 128), 256, 0, stream>>>(
        x, wd_hi, wd_lo, emb_hi, emb_lo, bias, out);
}
